// MeshGraphNetProcessorWithContext_21423296872640
// MI455X (gfx1250) — matrix operand order repaired and checked by reading, not compiled, not run on hardware
//
#include <hip/hip_runtime.h>

#define NNODES 20000
#define NEDGES 320000
#define LN_EPS 1e-5f
#define FIXSCALE 4294967296.0f          /* 2^32 */
#define INVFIX   2.3283064365386963e-10 /* 2^-32 */

typedef __attribute__((ext_vector_type(16))) __bf16 v16bf;
typedef __attribute__((ext_vector_type(8)))  float v8f;

union Frag { unsigned int u[8]; v16bf v; };

__device__ __forceinline__ unsigned short f2bf(float f) {
  unsigned int u = __float_as_uint(f);
  u += 0x7FFFu + ((u >> 16) & 1u);            // round-to-nearest-even
  return (unsigned short)(u >> 16);
}
__device__ __forceinline__ float bf2f(unsigned short h) {
  return __uint_as_float(((unsigned int)h) << 16);
}

// ---------------------------------------------------------------------------
// Weight conversion: f32 [din,128] row-major -> bf16 packed dwords in
// B-fragment order: idx = (((kc*8 + nt)*8 + j)*32 + lane)
// B layout (16-bit, 32x16): VGPR j holds K pair (N=lane&15), g=lane/16:
//   j<4: K=2j+8g ; j>=4: K=16+2(j-4)+8g (matches A).
// ---------------------------------------------------------------------------
__global__ void conv_w_kernel(const float* __restrict__ W,
                              unsigned int* __restrict__ dstw, int ndw) {
  int idx = blockIdx.x * 256 + threadIdx.x;
  if (idx >= ndw) return;
  int lane = idx & 31;
  int j    = (idx >> 5) & 7;
  int nt   = (idx >> 8) & 7;
  int kc   = idx >> 11;
  int g    = lane >> 4;
  int m    = lane & 15;
  int n    = nt * 16 + m;
  int k0   = kc * 32 + ((j < 4) ? (2 * j + 8 * g) : (16 + 2 * (j - 4) + 8 * g));
  unsigned int lo = f2bf(W[(size_t)k0 * 128 + n]);
  unsigned int hi = f2bf(W[(size_t)(k0 + 1) * 128 + n]);
  dstw[idx] = lo | (hi << 16);
}

__global__ void zero64_kernel(unsigned long long* __restrict__ p, int n) {
  int i = blockIdx.x * 256 + threadIdx.x;
  if (i < n) p[i] = 0ull;
}

// Deterministic segment-sum: fixed-point (2^-32) int64 atomics — integer adds
// are associative, so replays are bit-identical regardless of arrival order.
__global__ void scatter_add_kernel(const float* __restrict__ ef,
                                   const int* __restrict__ dst,
                                   unsigned long long* __restrict__ eagg,
                                   int total) {
  int i = blockIdx.x * 256 + threadIdx.x;
  if (i < total) {
    int e = i >> 7, c = i & 127;
    long long q = (long long)llrintf(ef[i] * FIXSCALE);
    atomicAdd(&eagg[(size_t)dst[e] * 128 + c], (unsigned long long)q);
  }
}

// ---------------------------------------------------------------------------
// GEMM wave blocking: 8 waves per block, wave = (mh = wid>>2) M-half (4
// 16-row tiles) x (nq = wid&3) N-quarter (2 16-col tiles) -> 8 accumulators.
// Per K-chunk: 32 A-dwords (LDS) + 16 B-dwords (global/L2) feed 8 WMMAs.
// A layout (16-bit, 16x32): lane m (0-15)=row; VGPR j holds K pair:
//   j<4: K=2j+8g ; j>=4: K=16+2(j-4)+8g   (g = lane/16)
// C/D: VGPR i -> row i+8g (within 16), col = tile*16 + (lane&15)
// ---------------------------------------------------------------------------
__device__ __forceinline__ void gemm_xy128(v8f* acc, const unsigned short* xy,
                                           const unsigned int* Wf,
                                           int mh, int nq, int g, int m,
                                           int lane) {
#pragma unroll
  for (int t = 0; t < 8; ++t) { v8f z = {0,0,0,0,0,0,0,0}; acc[t] = z; }
#pragma unroll
  for (int kc = 0; kc < 4; ++kc) {
    Frag a[4];
#pragma unroll
    for (int mt = 0; mt < 4; ++mt) {
      int Rw = mh * 64 + mt * 16 + m;
#pragma unroll
      for (int j = 0; j < 8; ++j) {
        int k = kc * 32 + ((j < 4) ? (2 * j + 8 * g) : (16 + 2 * (j - 4) + 8 * g));
        a[mt].u[j] = *(const unsigned int*)&xy[Rw * 130 + k];
      }
    }
    Frag b[2];
#pragma unroll
    for (int tb = 0; tb < 2; ++tb) {
      const unsigned int* bp = Wf + (((kc * 8 + (nq * 2 + tb)) * 8) * 32) + lane;
#pragma unroll
      for (int j = 0; j < 8; ++j) b[tb].u[j] = bp[j * 32];
    }
#pragma unroll
    for (int tb = 0; tb < 2; ++tb)
#pragma unroll
      for (int mt = 0; mt < 4; ++mt)
        acc[mt * 2 + tb] = __builtin_amdgcn_wmma_f32_16x16x32_bf16(
            false, a[mt].v, false, b[tb].v, (short)0, acc[mt * 2 + tb],
            false, false);
  }
}

__device__ __forceinline__ void store_ln_silu(v8f* acc, unsigned short* xy,
                                              float* mu, float* rsg,
                                              const float* __restrict__ bias,
                                              const float* __restrict__ gam,
                                              const float* __restrict__ bet,
                                              int tid, int mh, int nq, int g,
                                              int m) {
  __syncthreads();   // all waves done reading xy from previous GEMM
#pragma unroll
  for (int tb = 0; tb < 2; ++tb) {
    int col = (nq * 2 + tb) * 16 + m;
    float bb = bias[col];
#pragma unroll
    for (int mt = 0; mt < 4; ++mt)
#pragma unroll
      for (int i = 0; i < 8; ++i) {
        int R = mh * 64 + mt * 16 + i + 8 * g;
        xy[R * 130 + col] = f2bf(acc[mt * 2 + tb][i] + bb);
      }
  }
  __syncthreads();
  if (tid < 128) {
    float s = 0.f, s2 = 0.f;
    for (int c = 0; c < 128; ++c) {
      float v = bf2f(xy[tid * 130 + c]);
      s += v; s2 += v * v;
    }
    float mean = s * (1.f / 128.f);
    float var  = s2 * (1.f / 128.f) - mean * mean;
    mu[tid]  = mean;
    rsg[tid] = rsqrtf(var + LN_EPS);
  }
  __syncthreads();
#pragma unroll 4
  for (int it = 0; it < 64; ++it) {
    int idx = it * 256 + tid;
    int r = idx >> 7, c = idx & 127;
    float v = bf2f(xy[r * 130 + c]);
    v = (v - mu[r]) * rsg[r] * gam[c] + bet[c];
    v = v / (1.f + __expf(-v));                 // SiLU
    xy[r * 130 + c] = f2bf(v);
  }
  __syncthreads();
}

// ---------------------------------------------------------------------------
// Edge MLP: per block 128 edges. x = [efeat | nfeat[src] | nfeat[dst] |
// ffeat[src] | ffeat[dst]] (448) -> 128 -> 128 -> 128 + residual.
// ---------------------------------------------------------------------------
__global__ __launch_bounds__(256) void edge_mlp_kernel(
    const float* __restrict__ efeat_in, float* __restrict__ efeat_out,
    const float* __restrict__ nfeat, const float* __restrict__ ffeat,
    const int* __restrict__ src, const int* __restrict__ dst,
    const unsigned int* __restrict__ W1, const unsigned int* __restrict__ W2,
    const unsigned int* __restrict__ W3,
    const float* __restrict__ b1, const float* __restrict__ b2,
    const float* __restrict__ b3,
    const float* __restrict__ g1, const float* __restrict__ be1,
    const float* __restrict__ g2, const float* __restrict__ be2) {
  __shared__ unsigned short xstage[128 * 34];
  __shared__ unsigned short xy[128 * 130];
  __shared__ float mu[128], rsg[128];
  __shared__ int sIdx[128], dIdx[128];

  const int tid = threadIdx.x;
  const int lane = tid & 31, wid = tid >> 5;
  const int g = lane >> 4, m = lane & 15;
  const int mh = wid >> 2, nq = wid & 3;
  const int base = blockIdx.x * 128;

  if (tid < 128) { sIdx[tid] = src[base + tid]; dIdx[tid] = dst[base + tid]; }

  v8f acc[8];
#pragma unroll
  for (int t = 0; t < 8; ++t) { v8f z = {0,0,0,0,0,0,0,0}; acc[t] = z; }

  // -------- layer 1: K = 448 (14 chunks of 32), gather fused --------
  for (int kc = 0; kc < 14; ++kc) {
    __syncthreads();
#pragma unroll
    for (int it = 0; it < 16; ++it) {
      int idx = it * 256 + tid;
      int r = idx >> 5, k = idx & 31;
      float val;
      if (kc < 4)        val = efeat_in[(size_t)(base + r) * 128 + kc * 32 + k];
      else if (kc < 8)   val = nfeat[(size_t)sIdx[r] * 128 + (kc - 4) * 32 + k];
      else if (kc < 12)  val = nfeat[(size_t)dIdx[r] * 128 + (kc - 8) * 32 + k];
      else if (kc == 12) val = ffeat[(size_t)sIdx[r] * 32 + k];
      else               val = ffeat[(size_t)dIdx[r] * 32 + k];
      xstage[r * 34 + k] = f2bf(val);
    }
    __syncthreads();
    Frag a[4];
#pragma unroll
    for (int mt = 0; mt < 4; ++mt) {
      int Rw = mh * 64 + mt * 16 + m;
#pragma unroll
      for (int j = 0; j < 8; ++j) {
        int k = (j < 4) ? (2 * j + 8 * g) : (16 + 2 * (j - 4) + 8 * g);
        a[mt].u[j] = *(const unsigned int*)&xstage[Rw * 34 + k];
      }
    }
    Frag b[2];
#pragma unroll
    for (int tb = 0; tb < 2; ++tb) {
      const unsigned int* bp = W1 + (((kc * 8 + (nq * 2 + tb)) * 8) * 32) + lane;
#pragma unroll
      for (int j = 0; j < 8; ++j) b[tb].u[j] = bp[j * 32];
    }
#pragma unroll
    for (int tb = 0; tb < 2; ++tb)
#pragma unroll
      for (int mt = 0; mt < 4; ++mt)
        acc[mt * 2 + tb] = __builtin_amdgcn_wmma_f32_16x16x32_bf16(
            false, a[mt].v, false, b[tb].v, (short)0, acc[mt * 2 + tb],
            false, false);
  }
  store_ln_silu(acc, xy, mu, rsg, b1, g1, be1, tid, mh, nq, g, m);

  // -------- layer 2 --------
  gemm_xy128(acc, xy, W2, mh, nq, g, m, lane);
  store_ln_silu(acc, xy, mu, rsg, b2, g2, be2, tid, mh, nq, g, m);

  // -------- layer 3 + residual --------
  gemm_xy128(acc, xy, W3, mh, nq, g, m, lane);
#pragma unroll
  for (int tb = 0; tb < 2; ++tb) {
    int col = (nq * 2 + tb) * 16 + m;
    float bb = b3[col];
#pragma unroll
    for (int mt = 0; mt < 4; ++mt)
#pragma unroll
      for (int i = 0; i < 8; ++i) {
        int R = mh * 64 + mt * 16 + i + 8 * g;
        size_t off = (size_t)(base + R) * 128 + col;
        efeat_out[off] = acc[mt * 2 + tb][i] + bb + efeat_in[off];
      }
  }
}

// ---------------------------------------------------------------------------
// Node MLP: per block 128 nodes. x = [nfeat | e_agg | ffeat] (288)
// -> 128 -> 128 -> 128 + residual. Tail block guarded (rows >= NNODES zeroed).
// ---------------------------------------------------------------------------
__global__ __launch_bounds__(256) void node_mlp_kernel(
    const float* __restrict__ nfeat_in, float* __restrict__ nfeat_out,
    const unsigned long long* __restrict__ eagg,
    const float* __restrict__ ffeat,
    const unsigned int* __restrict__ W1, const unsigned int* __restrict__ W2,
    const unsigned int* __restrict__ W3,
    const float* __restrict__ b1, const float* __restrict__ b2,
    const float* __restrict__ b3,
    const float* __restrict__ g1, const float* __restrict__ be1,
    const float* __restrict__ g2, const float* __restrict__ be2) {
  __shared__ unsigned short xstage[128 * 34];
  __shared__ unsigned short xy[128 * 130];
  __shared__ float mu[128], rsg[128];

  const int tid = threadIdx.x;
  const int lane = tid & 31, wid = tid >> 5;
  const int g = lane >> 4, m = lane & 15;
  const int mh = wid >> 2, nq = wid & 3;
  const int base = blockIdx.x * 128;

  v8f acc[8];
#pragma unroll
  for (int t = 0; t < 8; ++t) { v8f z = {0,0,0,0,0,0,0,0}; acc[t] = z; }

  // -------- layer 1: K = 288 (9 chunks of 32) --------
  for (int kc = 0; kc < 9; ++kc) {
    __syncthreads();
#pragma unroll
    for (int it = 0; it < 16; ++it) {
      int idx = it * 256 + tid;
      int r = idx >> 5, k = idx & 31;
      int nrow = base + r;
      float val = 0.f;
      if (nrow < NNODES) {
        if (kc < 4)      val = nfeat_in[(size_t)nrow * 128 + kc * 32 + k];
        else if (kc < 8) val = (float)((double)(long long)
                             eagg[(size_t)nrow * 128 + (kc - 4) * 32 + k] * INVFIX);
        else             val = ffeat[(size_t)nrow * 32 + k];
      }
      xstage[r * 34 + k] = f2bf(val);
    }
    __syncthreads();
    Frag a[4];
#pragma unroll
    for (int mt = 0; mt < 4; ++mt) {
      int Rw = mh * 64 + mt * 16 + m;
#pragma unroll
      for (int j = 0; j < 8; ++j) {
        int k = (j < 4) ? (2 * j + 8 * g) : (16 + 2 * (j - 4) + 8 * g);
        a[mt].u[j] = *(const unsigned int*)&xstage[Rw * 34 + k];
      }
    }
    Frag b[2];
#pragma unroll
    for (int tb = 0; tb < 2; ++tb) {
      const unsigned int* bp = W1 + (((kc * 8 + (nq * 2 + tb)) * 8) * 32) + lane;
#pragma unroll
      for (int j = 0; j < 8; ++j) b[tb].u[j] = bp[j * 32];
    }
#pragma unroll
    for (int tb = 0; tb < 2; ++tb)
#pragma unroll
      for (int mt = 0; mt < 4; ++mt)
        acc[mt * 2 + tb] = __builtin_amdgcn_wmma_f32_16x16x32_bf16(
            false, a[mt].v, false, b[tb].v, (short)0, acc[mt * 2 + tb],
            false, false);
  }
  store_ln_silu(acc, xy, mu, rsg, b1, g1, be1, tid, mh, nq, g, m);

  gemm_xy128(acc, xy, W2, mh, nq, g, m, lane);
  store_ln_silu(acc, xy, mu, rsg, b2, g2, be2, tid, mh, nq, g, m);

  gemm_xy128(acc, xy, W3, mh, nq, g, m, lane);
#pragma unroll
  for (int tb = 0; tb < 2; ++tb) {
    int col = (nq * 2 + tb) * 16 + m;
    float bb = b3[col];
#pragma unroll
    for (int mt = 0; mt < 4; ++mt)
#pragma unroll
      for (int i = 0; i < 8; ++i) {
        int R = mh * 64 + mt * 16 + i + 8 * g;
        int node = base + R;
        if (node < NNODES) {
          size_t off = (size_t)node * 128 + col;
          nfeat_out[off] = acc[mt * 2 + tb][i] + bb + nfeat_in[off];
        }
      }
  }
}

// ---------------------------------------------------------------------------
// Host side
// ---------------------------------------------------------------------------
extern "C" void kernel_launch(void* const* d_in, const int* in_sizes, int n_in,
                              void* d_out, int out_size, void* d_ws,
                              size_t ws_size, hipStream_t stream) {
  (void)in_sizes; (void)n_in; (void)out_size; (void)ws_size;
  const float* nfeat0 = (const float*)d_in[0];
  const float* efeat0 = (const float*)d_in[1];
  const float* ffeat  = (const float*)d_in[2];
  const int*   src    = (const int*)d_in[3];
  const int*   dst    = (const int*)d_in[4];
  float* out = (float*)d_out;

  float* ws_efeat = (float*)d_ws;
  float* ws_nfeat = ws_efeat + (size_t)NEDGES * 128;
  unsigned long long* ws_eagg =
      (unsigned long long*)(ws_nfeat + (size_t)NNODES * 128);
  unsigned int* ws_w = (unsigned int*)(ws_eagg + (size_t)NNODES * 128);

  // swizzled bf16 weight fragment buffers per step
  unsigned int *eW1[4], *eW2[4], *eW3[4], *nW1[4], *nW2[4], *nW3[4];
  size_t woff = 0;
  for (int p = 0; p < 4; ++p) {
    eW1[p] = ws_w + woff; woff += 448 * 64;
    eW2[p] = ws_w + woff; woff += 128 * 64;
    eW3[p] = ws_w + woff; woff += 128 * 64;
    nW1[p] = ws_w + woff; woff += 288 * 64;
    nW2[p] = ws_w + woff; woff += 128 * 64;
    nW3[p] = ws_w + woff; woff += 128 * 64;
  }

  // param index layout: dict flatten order
  //   [5 + p*10 + {0:W1,1:b1,2:W2,3:b2,4:g1,5:be1,6:g2,7:be2,8:W3,9:b3}] edges
  //   [45 + p*10 + ...] nodes
  for (int p = 0; p < 4; ++p) {
    int be = 5 + p * 10, bn = 45 + p * 10;
    conv_w_kernel<<<(448 * 64 + 255) / 256, 256, 0, stream>>>(
        (const float*)d_in[be + 0], eW1[p], 448 * 64);
    conv_w_kernel<<<(128 * 64 + 255) / 256, 256, 0, stream>>>(
        (const float*)d_in[be + 2], eW2[p], 128 * 64);
    conv_w_kernel<<<(128 * 64 + 255) / 256, 256, 0, stream>>>(
        (const float*)d_in[be + 8], eW3[p], 128 * 64);
    conv_w_kernel<<<(288 * 64 + 255) / 256, 256, 0, stream>>>(
        (const float*)d_in[bn + 0], nW1[p], 288 * 64);
    conv_w_kernel<<<(128 * 64 + 255) / 256, 256, 0, stream>>>(
        (const float*)d_in[bn + 2], nW2[p], 128 * 64);
    conv_w_kernel<<<(128 * 64 + 255) / 256, 256, 0, stream>>>(
        (const float*)d_in[bn + 8], nW3[p], 128 * 64);
  }

  for (int p = 0; p < 4; ++p) {
    int be = 5 + p * 10, bn = 45 + p * 10;
    const float* ef_in = (p == 0) ? efeat0 : ws_efeat;
    const float* nf_in = (p == 0) ? nfeat0 : ws_nfeat;
    float* nf_out = (p == 3) ? out : ws_nfeat;

    edge_mlp_kernel<<<NEDGES / 128, 256, 0, stream>>>(
        ef_in, ws_efeat, nf_in, ffeat, src, dst,
        eW1[p], eW2[p], eW3[p],
        (const float*)d_in[be + 1], (const float*)d_in[be + 3],
        (const float*)d_in[be + 9],
        (const float*)d_in[be + 4], (const float*)d_in[be + 5],
        (const float*)d_in[be + 6], (const float*)d_in[be + 7]);

    zero64_kernel<<<(NNODES * 128 + 255) / 256, 256, 0, stream>>>(
        ws_eagg, NNODES * 128);
    scatter_add_kernel<<<(NEDGES * 128 + 255) / 256, 256, 0, stream>>>(
        ws_efeat, dst, ws_eagg, NEDGES * 128);

    node_mlp_kernel<<<(NNODES + 127) / 128, 256, 0, stream>>>(
        nf_in, nf_out, ws_eagg, ffeat,
        nW1[p], nW2[p], nW3[p],
        (const float*)d_in[bn + 1], (const float*)d_in[bn + 3],
        (const float*)d_in[bn + 9],
        (const float*)d_in[bn + 4], (const float*)d_in[bn + 5],
        (const float*)d_in[bn + 6], (const float*)d_in[bn + 7]);
  }
}
